// TimeAttentionGRU_19000935318220
// MI455X (gfx1250) — matrix operand order repaired and checked by reading, not compiled, not run on hardware
//
#include <hip/hip_runtime.h>
#include <hip/hip_bf16.h>

// ---------------------------------------------------------------------------
// Bidirectional lite-GRU + additive attention for gfx1250 (MI455X).
// All GEMMs use v_wmma_f32_16x16x32_bf16 (bf16 operands, f32 accum).
// Recurrent scan keeps Uz^T/Uh^T (bf16) + h (bf16) entirely in LDS
// (~272KB of the 320KB WGP LDS), h kept in f32 registers per wave.
// bf16 conversion via native casts -> v_cvt_pk_bf16_f32 (no ALU emulation).
// ---------------------------------------------------------------------------

constexpr int NB = 64;    // batch
constexpr int NS = 1024;  // sequence
constexpr int ND = 128;   // input dim
constexpr int NH = 256;   // hidden dim
constexpr int NH2 = 512;  // 2*H

typedef __attribute__((ext_vector_type(16))) __bf16 v16bf;
typedef __attribute__((ext_vector_type(8)))  float  v8f;
typedef __attribute__((ext_vector_type(4)))  unsigned int u32x4;

union ABFrag {
    v16bf v;
    u32x4 q[2];
    __bf16 h[16];
};

static __device__ __forceinline__ __bf16 tobf(float f) { return (__bf16)f; }

static __device__ __forceinline__ v8f wmma_bf16(v16bf a, v16bf b, v8f c) {
    // D(16x16,f32) = A(16x32,bf16) * B(32x16,bf16) + C
    return __builtin_amdgcn_wmma_f32_16x16x32_bf16(
        /*neg_a=*/false, a, /*neg_b=*/false, b,
        /*c_mod=*/(short)0, c, /*reuse_a=*/false, /*reuse_b=*/false);
}

// ---------------------------------------------------------------------------
// Kernel 1: input projections  proj[g][s][b][h] = x[b,s,:] @ W_g + bias_g
// One WG = 256 thr (8 waves) handles 64 rows x all 256 cols for one gate g.
// grid = (B*S/64, 4)
// ---------------------------------------------------------------------------
__global__ __launch_bounds__(256) void proj_kernel(
        const float* __restrict__ x,
        const float* __restrict__ Wzf, const float* __restrict__ bzf,
        const float* __restrict__ Whf, const float* __restrict__ bhf,
        const float* __restrict__ Wzb, const float* __restrict__ bzb,
        const float* __restrict__ Whb, const float* __restrict__ bhb,
        float* __restrict__ proj) {
    extern __shared__ __bf16 smw[];                  // W^T bf16 [NH][ND+8]
    constexpr int KP = ND + 8;

    const int gate = blockIdx.y;
    const float* W; const float* bias;
    switch (gate) {
        case 0:  W = Wzf; bias = bzf; break;
        case 1:  W = Whf; bias = bhf; break;
        case 2:  W = Wzb; bias = bzb; break;
        default: W = Whb; bias = bhb; break;
    }
    float* dst = proj + (size_t)gate * NS * NB * NH;

    const int tid = threadIdx.x;
    for (int i = tid; i < ND * NH; i += 256) {
        int k = i & (ND - 1);
        int n = i >> 7;
        smw[n * KP + k] = tobf(W[(size_t)k * NH + n]);
    }
    __syncthreads();

    const int wave = tid >> 5, lane = tid & 31;
    const int lrow = lane & 15;
    const int cA0 = (lane < 16) ? 0 : 8;             // A-frag K chunk base
    const int cB0 = (lane < 16) ? 0 : 8;             // B-frag K chunk base
    const int rowBlk = blockIdx.x * 64;              // flattened (b*NS + s)

    for (int mt = 0; mt < 4; ++mt) {
        const int arow = rowBlk + mt * 16 + lrow;
        const float* xr = x + (size_t)arow * ND;
        ABFrag a[4];
        #pragma unroll
        for (int k = 0; k < 4; ++k) {
            #pragma unroll
            for (int e = 0; e < 8; ++e) a[k].h[e]     = tobf(xr[32 * k + cA0 + e]);
            #pragma unroll
            for (int e = 0; e < 8; ++e) a[k].h[8 + e] = tobf(xr[32 * k + cA0 + 16 + e]);
        }
        #pragma unroll
        for (int t2 = 0; t2 < 2; ++t2) {
            const int nt = wave * 2 + t2;
            v8f acc = {};
            #pragma unroll
            for (int k = 0; k < 4; ++k) {
                const __bf16* src = &smw[(nt * 16 + lrow) * KP + 32 * k + cB0];
                ABFrag bf;
                bf.q[0] = *(const u32x4*)(src);
                bf.q[1] = *(const u32x4*)(src + 16);
                acc = wmma_bf16(a[k].v, bf.v, acc);
            }
            const int col = nt * 16 + lrow;
            const float bv = bias[col];
            #pragma unroll
            for (int r = 0; r < 8; ++r) {
                int grow = rowBlk + mt * 16 + r + ((lane < 16) ? 0 : 8);
                int bb = grow >> 10;                  // / NS
                int ss = grow & (NS - 1);
                dst[((size_t)ss * NB + bb) * NH + col] = acc[r] + bv;
            }
        }
    }
}

// ---------------------------------------------------------------------------
// Kernel 2: recurrent scan. grid = (4 batch-chunks, 2 directions), 512 thr.
// LDS: UzT[256][264] bf16 + UhT[256][264] bf16 + h_bf16[16][264].
// Wave w owns N-tile w of both gates; h kept f32 in registers across steps.
// Emits both f32 and bf16 copies of the output states.
// ---------------------------------------------------------------------------
__global__ __launch_bounds__(512) void gru_scan_kernel(
        const float* __restrict__ Uzf, const float* __restrict__ Uhf,
        const float* __restrict__ Uzb, const float* __restrict__ Uhb,
        const float* __restrict__ proj,
        float* __restrict__ out,            // [B,S,2H] f32
        __bf16* __restrict__ outbf) {       // [B,S,2H] bf16
    extern __shared__ __bf16 smg[];
    constexpr int KP = NH + 8;                        // 264, pad kills bank conflicts
    __bf16* UzT = smg;
    __bf16* UhT = smg + NH * KP;
    __bf16* hS  = smg + 2 * NH * KP;                  // [16][KP]

    const int dir = blockIdx.y;
    const int b0  = blockIdx.x * 16;
    const float* Uz = dir ? Uzb : Uzf;
    const float* Uh = dir ? Uhb : Uhf;
    const float* pz = proj + (size_t)(dir ? 2 : 0) * NS * NB * NH;
    const float* ph = proj + (size_t)(dir ? 3 : 1) * NS * NB * NH;

    const int tid = threadIdx.x;                      // 512
    for (int i = tid; i < NH * NH; i += 512) {        // transpose+convert U -> LDS
        int n = i >> 8, k = i & (NH - 1);
        UzT[n * KP + k] = tobf(Uz[(size_t)k * NH + n]);
        UhT[n * KP + k] = tobf(Uh[(size_t)k * NH + n]);
    }
    for (int i = tid; i < 16 * KP; i += 512) hS[i] = (__bf16)0.0f;
    __syncthreads();

    const int wave = tid >> 5, lane = tid & 31;       // wave == N-tile (0..15)
    const int lrow = lane & 15;
    const int cA0 = (lane < 16) ? 0 : 16;
    const int cB0 = (lane < 16) ? 0 : 16;
    const int col = wave * 16 + lrow;                 // output column
    const int mlo = (lane < 16) ? 0 : 8;
    const __bf16* hrow = &hS[lrow * KP];
    const __bf16* uzr  = &UzT[col * KP];
    const __bf16* uhr  = &UhT[col * KP];

    float hold[8];
    #pragma unroll
    for (int r = 0; r < 8; ++r) hold[r] = 0.f;

    for (int s = 0; s < NS; ++s) {
        const int t = dir ? (NS - 1 - s) : s;
        __syncthreads();                              // prev-step hS writes visible
        ABFrag a[8];
        #pragma unroll
        for (int k = 0; k < 8; ++k) {
            a[k].q[0] = *(const u32x4*)(hrow + 32 * k + cA0);
            a[k].q[1] = *(const u32x4*)(hrow + 32 * k + cA0 + 8);
        }
        __syncthreads();                              // all hS reads complete

        v8f accz = {}, acch = {};
        #pragma unroll
        for (int k = 0; k < 8; ++k) {
            ABFrag bz, bh;
            bz.q[0] = *(const u32x4*)(uzr + 32 * k + cB0);
            bz.q[1] = *(const u32x4*)(uzr + 32 * k + cB0 + 8);
            accz = wmma_bf16(a[k].v, bz.v, accz);
            bh.q[0] = *(const u32x4*)(uhr + 32 * k + cB0);
            bh.q[1] = *(const u32x4*)(uhr + 32 * k + cB0 + 8);
            acch = wmma_bf16(a[k].v, bh.v, acch);
        }

        #pragma unroll
        for (int r = 0; r < 8; ++r) {
            const int m = r + mlo;
            const size_t pidx = ((size_t)t * NB + (b0 + m)) * NH + col;
            float z  = 1.f / (1.f + __expf(-(accz[r] + pz[pidx])));
            float hc = tanhf(acch[r] + ph[pidx]);
            float v  = z * hold[r] + (1.f - z) * hc;
            hold[r] = v;
            const size_t oidx = ((size_t)(b0 + m) * NS + t) * NH2 + dir * NH + col;
            out[oidx]   = v;
            __bf16 vb   = tobf(v);
            outbf[oidx] = vb;
            hS[m * KP + col] = vb;
        }
    }
}

// ---------------------------------------------------------------------------
// Kernel 3a: W1 [512,512] f32 -> W1^T bf16 [512][512] (N-major for B-frags)
// ---------------------------------------------------------------------------
__global__ __launch_bounds__(256) void w1t_kernel(
        const float* __restrict__ W1, __bf16* __restrict__ W1T) {
    int i = blockIdx.x * blockDim.x + threadIdx.x;
    if (i < NH2 * NH2) {
        int n = i >> 9, k = i & (NH2 - 1);
        W1T[(size_t)n * NH2 + k] = tobf(W1[(size_t)k * NH2 + n]);
    }
}

// ---------------------------------------------------------------------------
// Kernel 3b: scores[b,s] = sum_j tanh(out_row . W1[:,j] + b1[j]) * w2[j]
// (b2 dropped: constant offset cancels in softmax). WMMA GEMM, K=512.
// One WG = 512 thr handles 32 rows; wave(w): Mtile=w>>3, Ngroup=w&7 (4 tiles).
// A operand loaded directly as bf16 (no conversion). scores pre-zeroed.
// ---------------------------------------------------------------------------
__global__ __launch_bounds__(512) void attn_score_kernel(
        const __bf16* __restrict__ outbf,   // [B*S][512] bf16
        const __bf16* __restrict__ W1T,     // [512][512] bf16 (N-major)
        const float* __restrict__ b1,
        const float* __restrict__ w2,
        float* __restrict__ scores) {
    const int tid = threadIdx.x, wave = tid >> 5, lane = tid & 31;
    const int lrow = lane & 15;
    const int cA0 = (lane < 16) ? 0 : 16;
    const int cB0 = (lane < 16) ? 0 : 16;
    const int mt = wave >> 3, ng = wave & 7;          // 2 M-tiles x 8 N-groups
    const int rowBase = blockIdx.x * 32 + mt * 16;    // flattened b*NS+s
    const __bf16* xr = outbf + (size_t)(rowBase + lrow) * NH2;

    v8f acc[4];
    v8f zero = {};
    #pragma unroll
    for (int j = 0; j < 4; ++j) acc[j] = zero;

    for (int k = 0; k < 16; ++k) {                    // K = 512
        ABFrag a;
        a.q[0] = *(const u32x4*)(xr + 32 * k + cA0);
        a.q[1] = *(const u32x4*)(xr + 32 * k + cA0 + 8);
        #pragma unroll
        for (int j = 0; j < 4; ++j) {
            const int n = (ng * 4 + j) * 16 + lrow;
            const __bf16* wr = W1T + (size_t)n * NH2 + 32 * k + cB0;
            ABFrag bf;
            bf.q[0] = *(const u32x4*)(wr);
            bf.q[1] = *(const u32x4*)(wr + 8);
            acc[j] = wmma_bf16(a.v, bf.v, acc[j]);
        }
    }

    float ps[8];
    #pragma unroll
    for (int r = 0; r < 8; ++r) ps[r] = 0.f;
    #pragma unroll
    for (int j = 0; j < 4; ++j) {
        const int colj = (ng * 4 + j) * 16 + lrow;
        const float b1v = b1[colj], w2v = w2[colj];
        #pragma unroll
        for (int r = 0; r < 8; ++r)
            ps[r] += tanhf(acc[j][r] + b1v) * w2v;
    }
    #pragma unroll
    for (int r = 0; r < 8; ++r) {                     // reduce over 16 cols (lanes)
        float v = ps[r];
        #pragma unroll
        for (int off = 1; off < 16; off <<= 1) v += __shfl_xor(v, off, 32);
        ps[r] = v;
    }
    if (lrow == 0) {
        #pragma unroll
        for (int r = 0; r < 8; ++r) {
            int row = rowBase + r + ((lane < 16) ? 0 : 8);
            atomicAdd(&scores[row], ps[r]);
        }
    }
}

// ---------------------------------------------------------------------------
// Kernel 4: softmax over S + context = sum_s attn * out.  grid = B, 256 thr.
// ---------------------------------------------------------------------------
__global__ __launch_bounds__(256) void softmax_ctx_kernel(
        const float* __restrict__ scores,
        const float* __restrict__ out,
        float* __restrict__ ctx) {
    __shared__ float red[256];
    __shared__ float wts[NS];
    const int b = blockIdx.x, tid = threadIdx.x;
    const float* sc = scores + (size_t)b * NS;

    float mx = -1e30f;
    for (int s = tid; s < NS; s += 256) mx = fmaxf(mx, sc[s]);
    red[tid] = mx; __syncthreads();
    for (int w = 128; w > 0; w >>= 1) {
        if (tid < w) red[tid] = fmaxf(red[tid], red[tid + w]);
        __syncthreads();
    }
    mx = red[0]; __syncthreads();

    float sum = 0.f;
    for (int s = tid; s < NS; s += 256) {
        float e = __expf(sc[s] - mx);
        wts[s] = e; sum += e;
    }
    red[tid] = sum; __syncthreads();
    for (int w = 128; w > 0; w >>= 1) {
        if (tid < w) red[tid] += red[tid + w];
        __syncthreads();
    }
    const float inv = 1.f / red[0];
    __syncthreads();

    for (int c = tid; c < NH2; c += 256) {
        const float* op = out + (size_t)b * NS * NH2 + c;
        float accv = 0.f;
        for (int s = 0; s < NS; ++s) accv += wts[s] * op[(size_t)s * NH2];
        ctx[(size_t)b * NH2 + c] = accv * inv;
    }
}

// ---------------------------------------------------------------------------
// Host-side orchestration
// ---------------------------------------------------------------------------
extern "C" void kernel_launch(void* const* d_in, const int* in_sizes, int n_in,
                              void* d_out, int out_size, void* d_ws, size_t ws_size,
                              hipStream_t stream) {
    (void)in_sizes; (void)n_in; (void)out_size; (void)ws_size;

    const float* x    = (const float*)d_in[0];
    const float* Wz_f = (const float*)d_in[1];
    const float* Uz_f = (const float*)d_in[2];
    const float* bz_f = (const float*)d_in[3];
    const float* Wh_f = (const float*)d_in[4];
    const float* Uh_f = (const float*)d_in[5];
    const float* bh_f = (const float*)d_in[6];
    const float* Wz_b = (const float*)d_in[7];
    const float* Uz_b = (const float*)d_in[8];
    const float* bz_b = (const float*)d_in[9];
    const float* Wh_b = (const float*)d_in[10];
    const float* Uh_b = (const float*)d_in[11];
    const float* bh_b = (const float*)d_in[12];
    const float* W1   = (const float*)d_in[13];
    const float* b1   = (const float*)d_in[14];
    const float* w2   = (const float*)d_in[15];
    // d_in[16] = b2: constant offset, cancels in softmax.

    // workspace carve (~449 MB)
    char* ws = (char*)d_ws;
    size_t off = 0;
    float*  proj   = (float*)(ws + off);  off += (size_t)4 * NS * NB * NH * sizeof(float);    // 256 MB
    float*  outbuf = (float*)(ws + off);  off += (size_t)NB * NS * NH2 * sizeof(float);       // 128 MB
    __bf16* outbf  = (__bf16*)(ws + off); off += (size_t)NB * NS * NH2 * sizeof(__bf16);      //  64 MB
    __bf16* W1T    = (__bf16*)(ws + off); off += (size_t)NH2 * NH2 * sizeof(__bf16);          // 512 KB
    float*  scores = (float*)(ws + off);  off += (size_t)NB * NS * sizeof(float);             // 256 KB

    hipMemsetAsync(scores, 0, (size_t)NB * NS * sizeof(float), stream);

    // 1) input projections: 4 gates, WMMA GEMM [65536,128]x[128,256]
    {
        size_t lds = (size_t)NH * (ND + 8) * sizeof(__bf16);   // ~68 KB
        proj_kernel<<<dim3(NB * NS / 64, 4), 256, lds, stream>>>(
            x, Wz_f, bz_f, Wh_f, bh_f, Wz_b, bz_b, Wh_b, bh_b, proj);
    }

    // 2) W1 transpose/convert for attention B-fragments
    w1t_kernel<<<dim3((NH2 * NH2 + 255) / 256), 256, 0, stream>>>(W1, W1T);

    // 3) recurrent scan: U matrices resident in LDS (~272 KB of 320 KB WGP LDS)
    {
        size_t lds = (size_t)(2 * NH * (NH + 8) + 16 * (NH + 8)) * sizeof(__bf16);
        gru_scan_kernel<<<dim3(4, 2), 512, lds, stream>>>(
            Uz_f, Uh_f, Uz_b, Uh_b, proj, outbuf, outbf);
    }

    // 4) attention scores: WMMA GEMM [65536,512]x[512,512] fused tanh/w2-reduce
    attn_score_kernel<<<dim3(NB * NS / 32), 512, 0, stream>>>(
        outbf, W1T, b1, w2, scores);

    // 5) softmax over time + weighted context sum
    softmax_ctx_kernel<<<dim3(NB), 256, 0, stream>>>(scores, outbuf, (float*)d_out);
}
